// AutoGraphModel_algo_17514876633761
// MI455X (gfx1250) — matrix operand order repaired and checked by reading, not compiled, not run on hardware
//
#include <hip/hip_runtime.h>
#include <hip/hip_bf16.h>

typedef _Float16 v16h __attribute__((ext_vector_type(16)));
typedef _Float16 h8   __attribute__((ext_vector_type(8)));
typedef float    v8f  __attribute__((ext_vector_type(8)));

#define SEQ 12

// ---------------- utility ----------------
__global__ void k_fill(float* __restrict__ p, float v, int n) {
    int i = blockIdx.x * blockDim.x + threadIdx.x;
    if (i < n) p[i] = v;
}

// ---------------- degree counting (self-loops folded into init=1.0) ----------------
__global__ void k_deg(const int* __restrict__ src, const int* __restrict__ dst,
                      float* __restrict__ deg_in, float* __restrict__ deg_out, int E) {
    int e = blockIdx.x * blockDim.x + threadIdx.x;
    if (e >= E) return;
    atomicAdd(&deg_in[dst[e]], 1.0f);
    atomicAdd(&deg_out[src[e]], 1.0f);
}

// ns = deg_out^-0.5, nd = deg_in^-0.5 ; msg2 init = self-loop contribution h(i)*ns(i)
__global__ void k_norm_msg2(const float* __restrict__ deg_in, const float* __restrict__ deg_out,
                            float* __restrict__ ns, float* __restrict__ nd,
                            float* __restrict__ msg2, int N) {
    int i = blockIdx.x * blockDim.x + threadIdx.x;
    if (i >= N) return;
    float s = rsqrtf(deg_out[i]);
    float d = rsqrtf(deg_in[i]);
    ns[i] = s; nd[i] = d;
    msg2[i * 2 + 0] = deg_in[i] * s;
    msg2[i * 2 + 1] = deg_out[i] * s;
}

__global__ void k_edge_msg2(const int* __restrict__ src, const int* __restrict__ dst,
                            const float* __restrict__ deg_in, const float* __restrict__ deg_out,
                            const float* __restrict__ ns, float* __restrict__ msg2, int E) {
    int e = blockIdx.x * blockDim.x + threadIdx.x;
    if (e >= E) return;
    int s = src[e], d = dst[e];
    float w = ns[s];
    atomicAdd(&msg2[d * 2 + 0], deg_in[s] * w);
    atomicAdd(&msg2[d * 2 + 1], deg_out[s] * w);
}

__device__ __forceinline__ float eluf(float x) { return x > 0.0f ? x : expm1f(x); }

// h[i][j] = elu( (msg2*nd) @ gc1_w + gc1_b )   (K = 2)
__global__ void k_gconv1(const float* __restrict__ msg2, const float* __restrict__ nd,
                         const float* __restrict__ w, const float* __restrict__ b,
                         float* __restrict__ h, int N) {
    int t = blockIdx.x * blockDim.x + threadIdx.x;
    if (t >= N * 128) return;
    int i = t >> 7, j = t & 127;
    float sc = nd[i];
    float m0 = msg2[i * 2 + 0] * sc;
    float m1 = msg2[i * 2 + 1] * sc;
    h[t] = eluf(fmaf(m0, w[j], fmaf(m1, w[128 + j], b[j])));
}

// msg128 init = self-loop contribution h(i)*ns(i)
__global__ void k_selfmsg128(const float* __restrict__ h, const float* __restrict__ ns,
                             float* __restrict__ msg128, int N) {
    int t = blockIdx.x * blockDim.x + threadIdx.x;
    if (t >= N * 128) return;
    msg128[t] = h[t] * ns[t >> 7];
}

// one wave per edge; lane covers 4 features; gather row h[s], scatter-add into msg128[d]
__global__ void k_edge_msg128(const int* __restrict__ src, const int* __restrict__ dst,
                              const float* __restrict__ h, const float* __restrict__ ns,
                              float* __restrict__ msg128, int E) {
    int t = blockIdx.x * blockDim.x + threadIdx.x;
    int e = t >> 5;
    int lane = t & 31;
    if (e >= E) return;
    int s = src[e], d = dst[e];
    float w = ns[s];
    __builtin_prefetch(&msg128[(size_t)d * 128], 1, 3);   // global_prefetch_b8
    const float4* hv = (const float4*)(h + (size_t)s * 128);
    float4 v = hv[lane];
    float* out = msg128 + (size_t)d * 128 + lane * 4;
    atomicAdd(out + 0, v.x * w);
    atomicAdd(out + 1, v.y * w);
    atomicAdd(out + 2, v.z * w);
    atomicAdd(out + 3, v.w * w);
}

// a16[i][j] = f16( msg128[i][j] * nd[i] )
__global__ void k_convert(const float* __restrict__ msg128, const float* __restrict__ nd,
                          _Float16* __restrict__ a16, int N) {
    int t = blockIdx.x * blockDim.x + threadIdx.x;
    if (t >= N * 128) return;
    a16[t] = (_Float16)(msg128[t] * nd[t >> 7]);
}

// Pack gc2_w [128,128] f32 into WMMA B-fragment order (f16):
// fragment (ct, ks): 32 lanes x 16 halfs; lane -> col ct*16+(lane&15), K = ks*32+(lane>>4)*8+(j&7)+(j>>3)*16
__global__ void k_bfrag(const float* __restrict__ W, _Float16* __restrict__ Bf) {
    int t = blockIdx.x * blockDim.x + threadIdx.x;   // 8*4*32*16 = 16384
    if (t >= 8 * 4 * 32 * 16) return;
    int j    = t & 15;
    int lane = (t >> 4) & 31;
    int ks   = (t >> 9) & 3;
    int ct   = t >> 11;
    int col  = ct * 16 + (lane & 15);
    int k    = ks * 32 + (lane >> 4) * 8 + (j & 7) + (j >> 3) * 16;
    Bf[t] = (_Float16)W[k * 128 + col];
}

// WMMA GEMM: Out[N,128] = elu( A[N,128](f16) @ W(f16, pre-packed) + bias )
// 8 waves / block, each wave computes a 16x128 row stripe; 32 v_wmma per wave.
__global__ __launch_bounds__(256)
void k_gemm_wmma(const _Float16* __restrict__ A, const _Float16* __restrict__ Bf,
                 const float* __restrict__ bias, float* __restrict__ Out) {
    const int wave = threadIdx.x >> 5;
    const int lane = threadIdx.x & 31;
    const int r0   = (blockIdx.x * 8 + wave) * 16;
    const int row  = r0 + (lane & 15);
    const int koff = (lane >> 4) * 8;          // 16-bit A 16x32 layout (ISA 7.12.2)

    v16h afrag[4];
    #pragma unroll
    for (int ks = 0; ks < 4; ++ks) {
        const _Float16* p = A + (size_t)row * 128 + ks * 32 + koff;
        h8 lo = *(const h8*)(p);
        h8 hi = *(const h8*)(p + 16);
        v16h a;
        #pragma unroll
        for (int j = 0; j < 8; ++j) { a[j] = lo[j]; a[j + 8] = hi[j]; }
        afrag[ks] = a;
    }
    #pragma unroll
    for (int ct = 0; ct < 8; ++ct) {
        v8f acc = {};
        #pragma unroll
        for (int ks = 0; ks < 4; ++ks) {
            const _Float16* bp = Bf + (size_t)((ct * 4 + ks) * 32 + lane) * 16;
            h8 blo = *(const h8*)(bp);
            h8 bhi = *(const h8*)(bp + 8);
            v16h b;
            #pragma unroll
            for (int j = 0; j < 8; ++j) { b[j] = blo[j]; b[j + 8] = bhi[j]; }
            acc = __builtin_amdgcn_wmma_f32_16x16x32_f16(
                false, afrag[ks], false, b, (short)0, acc, false, false);
        }
        const int col = ct * 16 + (lane & 15);
        const float bv = bias[col];
        #pragma unroll
        for (int v = 0; v < 8; ++v) {
            int rr = r0 + v + 8 * (lane >> 4);      // 32-bit C/D layout (ISA 7.12.2)
            Out[(size_t)rr * 128 + col] = eluf(acc[v] + bv);
        }
    }
}

// mean pool: graphs are contiguous blocks of P nodes
__global__ void k_pool(const float* __restrict__ h2, float* __restrict__ hg, int P) {
    int g = blockIdx.x, j = threadIdx.x;   // block = 128 threads
    const float* base = h2 + (size_t)g * P * 128 + j;
    float s = 0.0f;
    for (int p = 0; p < P; ++p) s += base[(size_t)p * 128];
    hg[g * 128 + j] = s / (float)P;
}

// generic small dense layer: Y[M,Nc] = act(X[M,K] @ W[K,Nc] + b)
__global__ void k_mlp(const float* __restrict__ X, const float* __restrict__ W,
                      const float* __restrict__ b, float* __restrict__ Y,
                      int M, int K, int Nc, int relu) {
    int t = blockIdx.x * blockDim.x + threadIdx.x;
    if (t >= M * Nc) return;
    int r = t / Nc, c = t % Nc;
    float s = b[c];
    const float* x = X + (size_t)r * K;
    for (int k = 0; k < K; ++k) s = fmaf(x[k], W[(size_t)k * Nc + c], s);
    Y[t] = relu ? fmaxf(s, 0.0f) : s;
}

__global__ void k_gather_algo(const int* __restrict__ ops, const float* __restrict__ e0,
                              const float* __restrict__ e1, const float* __restrict__ e2,
                              float* __restrict__ out, int B) {
    int t = blockIdx.x * blockDim.x + threadIdx.x;
    if (t >= B * 96) return;
    int b = t / 96, c = t % 96;
    float v;
    if (c < 32)      v = e0[ops[b * 3 + 0] * 32 + c];
    else if (c < 64) v = e1[ops[b * 3 + 1] * 32 + (c - 32)];
    else             v = e2[ops[b * 3 + 2] * 32 + (c - 64)];
    out[t] = v;
}

__global__ void k_gather_sch(const int* __restrict__ sch,
                             const float* __restrict__ e0, const float* __restrict__ e1,
                             const float* __restrict__ e2, const float* __restrict__ e3,
                             const float* __restrict__ e4, const float* __restrict__ e5,
                             float* __restrict__ out, int B) {
    int t = blockIdx.x * blockDim.x + threadIdx.x;
    if (t >= B * 192) return;
    int b = t / 192, c = t % 192;
    int slot = c >> 5, off = c & 31;
    const float* tabs[6] = {e0, e1, e2, e3, e4, e5};
    out[t] = tabs[slot][sch[b * 6 + slot] * 32 + off];
}

// fused 4-layer post-norm transformer encoder + 3-layer head, one block per batch elem
__global__ __launch_bounds__(128)
void k_transformer(const float* __restrict__ algo_feat, const float* __restrict__ graph_feat,
                   const float* __restrict__ sch_feat,
                   const float* __restrict__ t_wqkv, const float* __restrict__ t_bqkv,
                   const float* __restrict__ t_wo,   const float* __restrict__ t_bo,
                   const float* __restrict__ ln1g,   const float* __restrict__ ln1b,
                   const float* __restrict__ t_w1,   const float* __restrict__ t_b1,
                   const float* __restrict__ t_w2,   const float* __restrict__ t_b2,
                   const float* __restrict__ ln2g,   const float* __restrict__ ln2b,
                   const float* __restrict__ f_w1,   const float* __restrict__ f_b1,
                   const float* __restrict__ f_w2,   const float* __restrict__ f_b2,
                   const float* __restrict__ f_w3,   const float* __restrict__ f_b3,
                   float* __restrict__ out) {
    __shared__ float x[SEQ * 32];
    __shared__ float qkv[SEQ * 96];
    __shared__ float att[4 * SEQ * SEQ];
    __shared__ float o[SEQ * 32];
    __shared__ float f1[SEQ * 384];
    __shared__ float hh1[128];
    __shared__ float hh2[64];
    const int b = blockIdx.x, t = threadIdx.x;   // 128 threads

    for (int i = t; i < 384; i += 128) {
        float v;
        if (i < 128)      v = algo_feat[b * 128 + i];
        else if (i < 256) v = graph_feat[b * 128 + (i - 128)];
        else              v = sch_feat[b * 128 + (i - 256)];
        x[i] = v;
    }
    __syncthreads();

    for (int l = 0; l < 4; ++l) {
        const float* wqkv = t_wqkv + l * 32 * 96; const float* bqkv = t_bqkv + l * 96;
        for (int i = t; i < SEQ * 96; i += 128) {
            int r = i / 96, c = i % 96;
            float s = bqkv[c];
            for (int k = 0; k < 32; ++k) s = fmaf(x[r * 32 + k], wqkv[k * 96 + c], s);
            qkv[i] = s;
        }
        __syncthreads();
        const float scale = 0.35355339059327373f;   // 1/sqrt(8)
        if (t < 4 * SEQ) {
            int h = t / SEQ, q = t % SEQ;
            float sc[SEQ]; float mx = -3.4e38f;
            #pragma unroll
            for (int k2 = 0; k2 < SEQ; ++k2) {
                float s = 0.0f;
                #pragma unroll
                for (int d = 0; d < 8; ++d)
                    s = fmaf(qkv[q * 96 + h * 8 + d], qkv[k2 * 96 + 32 + h * 8 + d], s);
                s *= scale; sc[k2] = s; mx = fmaxf(mx, s);
            }
            float sum = 0.0f;
            #pragma unroll
            for (int k2 = 0; k2 < SEQ; ++k2) { float e = __expf(sc[k2] - mx); sc[k2] = e; sum += e; }
            float inv = 1.0f / sum;
            #pragma unroll
            for (int k2 = 0; k2 < SEQ; ++k2) att[(h * SEQ + q) * SEQ + k2] = sc[k2] * inv;
        }
        __syncthreads();
        for (int i = t; i < SEQ * 32; i += 128) {
            int q = i / 32, hd = i % 32, h = hd >> 3;
            float s = 0.0f;
            #pragma unroll
            for (int k2 = 0; k2 < SEQ; ++k2)
                s = fmaf(att[(h * SEQ + q) * SEQ + k2], qkv[k2 * 96 + 64 + hd], s);
            o[i] = s;
        }
        __syncthreads();
        const float* wo = t_wo + l * 32 * 32; const float* bo = t_bo + l * 32;
        for (int i = t; i < SEQ * 32; i += 128) {
            int q = i / 32, c = i % 32;
            float s = bo[c];
            for (int k = 0; k < 32; ++k) s = fmaf(o[q * 32 + k], wo[k * 32 + c], s);
            qkv[i] = x[i] + s;           // residual, staged in reused buffer
        }
        __syncthreads();
        if (t < SEQ) {
            float m = 0.0f;
            for (int c = 0; c < 32; ++c) m += qkv[t * 32 + c];
            m *= (1.0f / 32.0f);
            float v = 0.0f;
            for (int c = 0; c < 32; ++c) { float d = qkv[t * 32 + c] - m; v = fmaf(d, d, v); }
            float inv = rsqrtf(v * (1.0f / 32.0f) + 1e-5f);
            for (int c = 0; c < 32; ++c)
                x[t * 32 + c] = (qkv[t * 32 + c] - m) * inv * ln1g[l * 32 + c] + ln1b[l * 32 + c];
        }
        __syncthreads();
        const float* w1 = t_w1 + l * 32 * 384; const float* b1 = t_b1 + l * 384;
        for (int i = t; i < SEQ * 384; i += 128) {
            int r = i / 384, c = i % 384;
            float s = b1[c];
            for (int k = 0; k < 32; ++k) s = fmaf(x[r * 32 + k], w1[k * 384 + c], s);
            f1[i] = fmaxf(s, 0.0f);
        }
        __syncthreads();
        const float* w2 = t_w2 + l * 384 * 32; const float* b2 = t_b2 + l * 32;
        for (int i = t; i < SEQ * 32; i += 128) {
            int r = i / 32, c = i % 32;
            float s = b2[c];
            for (int k = 0; k < 384; ++k) s = fmaf(f1[r * 384 + k], w2[k * 32 + c], s);
            qkv[i] = x[i] + s;
        }
        __syncthreads();
        if (t < SEQ) {
            float m = 0.0f;
            for (int c = 0; c < 32; ++c) m += qkv[t * 32 + c];
            m *= (1.0f / 32.0f);
            float v = 0.0f;
            for (int c = 0; c < 32; ++c) { float d = qkv[t * 32 + c] - m; v = fmaf(d, d, v); }
            float inv = rsqrtf(v * (1.0f / 32.0f) + 1e-5f);
            for (int c = 0; c < 32; ++c)
                x[t * 32 + c] = (qkv[t * 32 + c] - m) * inv * ln2g[l * 32 + c] + ln2b[l * 32 + c];
        }
        __syncthreads();
    }
    // head: 384 -> 128 -> 64 -> 1
    {
        float s = f_b1[t];
        for (int k = 0; k < 384; ++k) s = fmaf(x[k], f_w1[k * 128 + t], s);
        hh1[t] = fmaxf(s, 0.0f);
    }
    __syncthreads();
    if (t < 64) {
        float s = f_b2[t];
        for (int k = 0; k < 128; ++k) s = fmaf(hh1[k], f_w2[k * 64 + t], s);
        hh2[t] = fmaxf(s, 0.0f);
    }
    __syncthreads();
    if (t == 0) {
        float s = f_b3[0];
        for (int k = 0; k < 64; ++k) s = fmaf(hh2[k], f_w3[k], s);
        out[b] = s;
    }
}

// ---------------- host ----------------
extern "C" void kernel_launch(void* const* d_in, const int* in_sizes, int n_in,
                              void* d_out, int out_size, void* d_ws, size_t ws_size,
                              hipStream_t stream) {
    const int*   algo_ops = (const int*)d_in[0];
    const int*   schedule = (const int*)d_in[1];
    const int*   src      = (const int*)d_in[2];
    const int*   dst      = (const int*)d_in[3];
    const float* emb_mc   = (const float*)d_in[5];
    const float* emb_mr   = (const float*)d_in[6];
    const float* emb_cm   = (const float*)d_in[7];
    const float* algo_w1  = (const float*)d_in[8];
    const float* algo_b1  = (const float*)d_in[9];
    const float* algo_w2  = (const float*)d_in[10];
    const float* algo_b2  = (const float*)d_in[11];
    const float* gc1_w    = (const float*)d_in[12];
    const float* gc1_b    = (const float*)d_in[13];
    const float* gc2_w    = (const float*)d_in[14];
    const float* gc2_b    = (const float*)d_in[15];
    const float* graph_w  = (const float*)d_in[16];
    const float* graph_b  = (const float*)d_in[17];
    const float* emb_kf   = (const float*)d_in[18];
    const float* emb_lb   = (const float*)d_in[19];
    const float* emb_dir  = (const float*)d_in[20];
    const float* emb_dep  = (const float*)d_in[21];
    const float* emb_fo   = (const float*)d_in[22];
    const float* emb_pr   = (const float*)d_in[23];
    const float* sch_w1   = (const float*)d_in[24];
    const float* sch_b1   = (const float*)d_in[25];
    const float* sch_w2   = (const float*)d_in[26];
    const float* sch_b2   = (const float*)d_in[27];
    const float* t_wqkv   = (const float*)d_in[28];
    const float* t_bqkv   = (const float*)d_in[29];
    const float* t_wo     = (const float*)d_in[30];
    const float* t_bo     = (const float*)d_in[31];
    const float* ln1g     = (const float*)d_in[32];
    const float* ln1b     = (const float*)d_in[33];
    const float* t_w1     = (const float*)d_in[34];
    const float* t_b1     = (const float*)d_in[35];
    const float* t_w2     = (const float*)d_in[36];
    const float* t_b2     = (const float*)d_in[37];
    const float* ln2g     = (const float*)d_in[38];
    const float* ln2b     = (const float*)d_in[39];
    const float* f_w1     = (const float*)d_in[40];
    const float* f_b1     = (const float*)d_in[41];
    const float* f_w2     = (const float*)d_in[42];
    const float* f_b2     = (const float*)d_in[43];
    const float* f_w3     = (const float*)d_in[44];
    const float* f_b3     = (const float*)d_in[45];

    const int E = in_sizes[2];
    const int N = in_sizes[4];
    const int B = in_sizes[0] / 3;
    const int P = N / B;

    // workspace bump allocator (256B aligned)
    char* w = (char*)d_ws;
    auto alloc = [&](size_t bytes) -> char* {
        char* p = w; w += (bytes + 255) & ~(size_t)255; return p;
    };
    float*    deg_in  = (float*)alloc((size_t)N * 4);
    float*    deg_out = (float*)alloc((size_t)N * 4);
    float*    ns      = (float*)alloc((size_t)N * 4);
    float*    nd      = (float*)alloc((size_t)N * 4);
    float*    msg2    = (float*)alloc((size_t)N * 2 * 4);
    float*    h       = (float*)alloc((size_t)N * 128 * 4);   // gconv1 out, reused for gconv2 out
    float*    msg128  = (float*)alloc((size_t)N * 128 * 4);
    _Float16* a16     = (_Float16*)alloc((size_t)N * 128 * 2);
    _Float16* bfrag   = (_Float16*)alloc((size_t)128 * 128 * 2);
    float*    algo_in = (float*)alloc((size_t)B * 96 * 4);
    float*    sch_in  = (float*)alloc((size_t)B * 192 * 4);
    float*    hid     = (float*)alloc((size_t)B * 128 * 4);
    float*    algo_f  = (float*)alloc((size_t)B * 128 * 4);
    float*    sch_f   = (float*)alloc((size_t)B * 128 * 4);
    float*    hg      = (float*)alloc((size_t)B * 128 * 4);
    float*    graph_f = (float*)alloc((size_t)B * 128 * 4);

    const int TB = 256;
    auto g1 = [&](long n) { return dim3((unsigned)((n + TB - 1) / TB)); };

    // --- graph branch ---
    k_fill<<<g1(N), TB, 0, stream>>>(deg_in, 1.0f, N);    // self loops
    k_fill<<<g1(N), TB, 0, stream>>>(deg_out, 1.0f, N);
    k_deg<<<g1(E), TB, 0, stream>>>(src, dst, deg_in, deg_out, E);
    k_norm_msg2<<<g1(N), TB, 0, stream>>>(deg_in, deg_out, ns, nd, msg2, N);
    k_edge_msg2<<<g1(E), TB, 0, stream>>>(src, dst, deg_in, deg_out, ns, msg2, E);
    k_gconv1<<<g1((long)N * 128), TB, 0, stream>>>(msg2, nd, gc1_w, gc1_b, h, N);
    k_selfmsg128<<<g1((long)N * 128), TB, 0, stream>>>(h, ns, msg128, N);
    k_edge_msg128<<<g1((long)E * 32), TB, 0, stream>>>(src, dst, h, ns, msg128, E);
    k_convert<<<g1((long)N * 128), TB, 0, stream>>>(msg128, nd, a16, N);
    k_bfrag<<<g1(16384), TB, 0, stream>>>(gc2_w, bfrag);
    k_gemm_wmma<<<dim3(N / 128), 256, 0, stream>>>(a16, bfrag, gc2_b, h);   // h := gconv2 out
    k_pool<<<dim3(B), 128, 0, stream>>>(h, hg, P);
    k_mlp<<<g1((long)B * 128), TB, 0, stream>>>(hg, graph_w, graph_b, graph_f, B, 128, 128, 0);

    // --- algo branch ---
    k_gather_algo<<<g1((long)B * 96), TB, 0, stream>>>(algo_ops, emb_mc, emb_mr, emb_cm, algo_in, B);
    k_mlp<<<g1((long)B * 128), TB, 0, stream>>>(algo_in, algo_w1, algo_b1, hid, B, 96, 128, 1);
    k_mlp<<<g1((long)B * 128), TB, 0, stream>>>(hid, algo_w2, algo_b2, algo_f, B, 128, 128, 0);

    // --- schedule branch ---
    k_gather_sch<<<g1((long)B * 192), TB, 0, stream>>>(schedule, emb_kf, emb_lb, emb_dir,
                                                       emb_dep, emb_fo, emb_pr, sch_in, B);
    k_mlp<<<g1((long)B * 128), TB, 0, stream>>>(sch_in, sch_w1, sch_b1, hid, B, 192, 128, 1);
    k_mlp<<<g1((long)B * 128), TB, 0, stream>>>(hid, sch_w2, sch_b2, sch_f, B, 128, 128, 0);

    // --- transformer + head ---
    k_transformer<<<dim3(B), 128, 0, stream>>>(algo_f, graph_f, sch_f,
                                               t_wqkv, t_bqkv, t_wo, t_bo, ln1g, ln1b,
                                               t_w1, t_b1, t_w2, t_b2, ln2g, ln2b,
                                               f_w1, f_b1, f_w2, f_b2, f_w3, f_b3,
                                               (float*)d_out);
}
